// AdditiveAttn_86818468922114
// MI455X (gfx1250) — matrix operand order repaired and checked by reading, not compiled, not run on hardware
//
#include <hip/hip_runtime.h>
#include <hip/hip_bf16.h>
#include <math.h>

// Problem sizes (fixed by reference)
#define BB   32
#define SS   4096
#define EE   512      // ENC == DEC == ATTN == 512
#define AA   512
#define TS   64       // rows of enc per workgroup in score kernel
#define LDSK 520      // padded LDS row stride (bf16 elements)

typedef __attribute__((ext_vector_type(16))) __bf16 v16bf;
typedef __attribute__((ext_vector_type(8)))  __bf16 v8bf;
typedef __attribute__((ext_vector_type(8)))  float  v8f;

__device__ __forceinline__ __bf16 f2bf(float f) { return (__bf16)f; }

__device__ __forceinline__ unsigned short f2bf_bits(float f) {
    __bf16 b = (__bf16)f;
    unsigned short u; __builtin_memcpy(&u, &b, 2); return u;
}

// Hardware transcendental tanh (V_TANH_F32) when available on gfx1250.
__device__ __forceinline__ float fast_tanh(float x) {
#if __has_builtin(__builtin_amdgcn_tanhf)
    return __builtin_amdgcn_tanhf(x);
#elif __has_builtin(__builtin_amdgcn_tanh_f32)
    return __builtin_amdgcn_tanh_f32(x);
#else
    return tanhf(x);
#endif
}

// ---------------------------------------------------------------------------
// K1: W_s [E][A] fp32  ->  WsT [A][E] bf16  (transposed so B fragments are
// contiguous 32-byte per-lane loads for v_wmma)
__global__ __launch_bounds__(256) void k_wst(const float* __restrict__ Ws,
                                             unsigned short* __restrict__ WsT) {
    int k = blockIdx.x;          // 0..511 (ENC index)
    int tid = threadIdx.x;       // n over two halves
    float x0 = Ws[(size_t)k * AA + tid];
    float x1 = Ws[(size_t)k * AA + tid + 256];
    WsT[(size_t)tid * EE + k]         = f2bf_bits(x0);
    WsT[(size_t)(tid + 256) * EE + k] = f2bf_bits(x1);
}

// ---------------------------------------------------------------------------
// K0: query = h @ W_h   ([32,512] x [512,512], tiny)
__global__ __launch_bounds__(128) void k_query(const float* __restrict__ h,
                                               const float* __restrict__ Wh,
                                               float* __restrict__ query) {
    __shared__ float hs[EE];
    int b = blockIdx.x, tid = threadIdx.x;
    ((float4*)hs)[tid] = ((const float4*)(h + (size_t)b * EE))[tid];
    __syncthreads();
    float acc0 = 0.f, acc1 = 0.f, acc2 = 0.f, acc3 = 0.f;
    for (int k = 0; k < EE; ++k) {
        float hv = hs[k];
        const float* w = Wh + (size_t)k * AA + tid;
        acc0 = fmaf(hv, w[0],   acc0);
        acc1 = fmaf(hv, w[128], acc1);
        acc2 = fmaf(hv, w[256], acc2);
        acc3 = fmaf(hv, w[384], acc3);
    }
    float* q = query + (size_t)b * AA + tid;
    q[0] = acc0; q[128] = acc1; q[256] = acc2; q[384] = acc3;
}

// ---------------------------------------------------------------------------
// KA: masked scores via bf16 WMMA.
//   grid = (S/TS, B), block = 256 (8 waves).
//   Each wave: N-slice of 64 (nt=4), M-tiles {0..3} (mt=4, 64 rows), K = 512
//   in 16 steps of 32.  Each B fragment feeds 4 WMMAs (good L2 reuse).
__global__ __launch_bounds__(256) void k_scores(const float* __restrict__ enc,
                                                const unsigned char* __restrict__ mask,
                                                const unsigned short* __restrict__ WsTu,
                                                const float* __restrict__ query,
                                                const float* __restrict__ vvec,
                                                float* __restrict__ scores) {
    __shared__ __bf16 Abf[TS][LDSK];
    __shared__ float  scAcc[8][TS];

    const __bf16* WsT = (const __bf16*)WsTu;
    const int b   = blockIdx.y;
    const int s0  = blockIdx.x * TS;
    const int tid = threadIdx.x;
    const int wave = tid >> 5;
    const int lane = tid & 31;
    const int lnn  = lane & 15;   // row of A-frag / col of B-frag
    const int hi   = lane >> 4;   // upper/lower 16-lane group

    // Stage enc tile [TS][512] fp32 -> bf16 LDS (32 float4 per thread)
    const float4* src = (const float4*)(enc + ((size_t)b * SS + s0) * EE);
    #pragma unroll
    for (int i = 0; i < (TS * EE / 4) / 256; ++i) {
        int idx  = tid + i * 256;         // float4 index within tile
        int flat = idx * 4;
        int r = flat >> 9;                // /512
        int k = flat & 511;
        float4 f = src[(size_t)r * (EE / 4) + (k >> 2)];
        __bf16* dst = &Abf[r][k];
        dst[0] = f2bf(f.x); dst[1] = f2bf(f.y);
        dst[2] = f2bf(f.z); dst[3] = f2bf(f.w);
    }
    __syncthreads();

    v8f acc[4][4];                        // [nt][mt]
    #pragma unroll
    for (int nt = 0; nt < 4; ++nt)
        #pragma unroll
        for (int mt = 0; mt < 4; ++mt)
            acc[nt][mt] = (v8f){};

    const int nbase = wave * 64 + lnn;

    for (int kc = 0; kc < 16; ++kc) {
        // A fragments in contiguous k order (same as B):
        //   a[j] = A[row][kc*32 + hi*16 + j]
        v16bf a[4];
        #pragma unroll
        for (int mt = 0; mt < 4; ++mt) {
            v8bf lo = *(const v8bf*)&Abf[mt * 16 + lnn][kc * 32 + hi * 16];
            v8bf hh = *(const v8bf*)&Abf[mt * 16 + lnn][kc * 32 + hi * 16 + 8];
            #pragma unroll
            for (int j = 0; j < 8; ++j) { a[mt][j] = lo[j]; a[mt][j + 8] = hh[j]; }
        }
        #pragma unroll
        for (int nt = 0; nt < 4; ++nt) {
            // B fragment: b[j] = B[K = kc*32 + hi*16 + j][n]; WsT is [n][k],
            // so this is 16 contiguous bf16 = one 32B-aligned vector load.
            const v16bf bm = *(const v16bf*)(WsT + (size_t)(nbase + nt * 16) * EE
                                                  + kc * 32 + hi * 16);
            #pragma unroll
            for (int mt = 0; mt < 4; ++mt)
                acc[nt][mt] = __builtin_amdgcn_wmma_f32_16x16x32_bf16(
                    false, a[mt], false, bm, (short)0, acc[nt][mt], false, false);
        }
    }

    // Per-lane: sum over this lane's N values of tanh(q + key) * v
    float qv[4], vw[4];
    #pragma unroll
    for (int nt = 0; nt < 4; ++nt) {
        int n = nbase + nt * 16;
        qv[nt] = query[(size_t)b * AA + n];
        vw[nt] = vvec[n];
    }
    #pragma unroll
    for (int mt = 0; mt < 4; ++mt) {
        #pragma unroll
        for (int vv = 0; vv < 8; ++vv) {   // C layout: M = mt*16 + hi*8 + vv
            float s = 0.f;
            #pragma unroll
            for (int nt = 0; nt < 4; ++nt)
                s += fast_tanh(qv[nt] + acc[nt][mt][vv]) * vw[nt];
            // reduce across the 16 lanes sharing this (mt, hi, vv)
            #pragma unroll
            for (int m = 1; m < 16; m <<= 1)
                s += __shfl_xor(s, m, 32);
            if (lnn == 0)
                scAcc[wave][mt * 16 + hi * 8 + vv] = s;   // per-wave partial
        }
    }
    __syncthreads();

    if (tid < TS) {
        float s = 0.f;
        #pragma unroll
        for (int w = 0; w < 8; ++w) s += scAcc[w][tid];   // fixed order: deterministic
        unsigned char mk = mask[(size_t)b * SS + s0 + tid];
        scores[(size_t)b * SS + s0 + tid] = mk ? -1e9f : s;
    }
}

// ---------------------------------------------------------------------------
// KB: softmax over S per batch row -> attn (into d_out)
__global__ __launch_bounds__(256) void k_softmax(const float* __restrict__ scores,
                                                 float* __restrict__ attn) {
    __shared__ float red[256];
    int b = blockIdx.x, tid = threadIdx.x;
    const float* sc = scores + (size_t)b * SS;
    float m = -3.4e38f;
    for (int i = tid; i < SS; i += 256) m = fmaxf(m, sc[i]);
    red[tid] = m; __syncthreads();
    for (int off = 128; off > 0; off >>= 1) {
        if (tid < off) red[tid] = fmaxf(red[tid], red[tid + off]);
        __syncthreads();
    }
    float M = red[0]; __syncthreads();
    float s = 0.f;
    for (int i = tid; i < SS; i += 256) s += __expf(sc[i] - M);
    red[tid] = s; __syncthreads();
    for (int off = 128; off > 0; off >>= 1) {
        if (tid < off) red[tid] += red[tid + off];
        __syncthreads();
    }
    float inv = 1.0f / red[0];
    for (int i = tid; i < SS; i += 256)
        attn[(size_t)b * SS + i] = __expf(sc[i] - M) * inv;
}

// ---------------------------------------------------------------------------
// KC: partial ctx over 256-row S-chunks.  grid = (16, B)
__global__ __launch_bounds__(256) void k_ctx_part(const float* __restrict__ enc,
                                                  const float* __restrict__ attn,
                                                  float* __restrict__ ctxp) {
    __shared__ float aw[256];
    int sc = blockIdx.x, b = blockIdx.y, tid = threadIdx.x;
    int sbase = sc * 256;
    aw[tid] = attn[(size_t)b * SS + sbase + tid];
    __syncthreads();
    float a0 = 0.f, a1 = 0.f;
    const float* ep = enc + ((size_t)b * SS + sbase) * EE;
    for (int s = 0; s < 256; ++s) {
        float w = aw[s];
        a0 = fmaf(w, ep[(size_t)s * EE + tid],       a0);
        a1 = fmaf(w, ep[(size_t)s * EE + tid + 256], a1);
    }
    float* out = ctxp + ((size_t)b * 16 + sc) * EE;
    out[tid] = a0; out[tid + 256] = a1;
}

// KD: fixed-order reduction of ctx partials -> d_out ctx
__global__ __launch_bounds__(256) void k_ctx_red(const float* __restrict__ ctxp,
                                                 float* __restrict__ ctx) {
    int b = blockIdx.x, tid = threadIdx.x;
    float a0 = 0.f, a1 = 0.f;
    for (int sc = 0; sc < 16; ++sc) {
        const float* p = ctxp + ((size_t)b * 16 + sc) * EE;
        a0 += p[tid]; a1 += p[tid + 256];
    }
    ctx[(size_t)b * EE + tid] = a0;
    ctx[(size_t)b * EE + tid + 256] = a1;
}

// ---------------------------------------------------------------------------
extern "C" void kernel_launch(void* const* d_in, const int* in_sizes, int n_in,
                              void* d_out, int out_size, void* d_ws, size_t ws_size,
                              hipStream_t stream) {
    const float*         h    = (const float*)d_in[0];          // [B, DEC]
    const float*         enc  = (const float*)d_in[1];          // [B, S, ENC]
    const unsigned char* mask = (const unsigned char*)d_in[2];  // [B, S] bool (1B)
    const float*         Wh   = (const float*)d_in[3];          // [DEC, ATTN]
    const float*         Ws   = (const float*)d_in[4];          // [ENC, ATTN]
    const float*         vv   = (const float*)d_in[5];          // [ATTN]

    float* out      = (float*)d_out;
    float* out_ctx  = out;                 // [B, ENC]
    float* out_attn = out + BB * EE;       // [B, S]

    char* ws = (char*)d_ws;
    unsigned short* WsT   = (unsigned short*)ws;          // 512 KB bf16 [A][E]
    float*          query = (float*)(ws + 524288);        // 64 KB
    float*          scores= (float*)(ws + 589824);        // 512 KB
    float*          ctxp  = (float*)(ws + 1114112);       // 1 MB  (ends ~2.06 MB)

    k_wst    <<<dim3(EE),          256, 0, stream>>>(Ws, WsT);
    k_query  <<<dim3(BB),          128, 0, stream>>>(h, Wh, query);
    k_scores <<<dim3(SS / TS, BB), 256, 0, stream>>>(enc, mask, WsT, query, vv, scores);
    k_softmax<<<dim3(BB),          256, 0, stream>>>(scores, out_attn);
    k_ctx_part<<<dim3(16, BB),     256, 0, stream>>>(enc, out_attn, ctxp);
    k_ctx_red<<<dim3(BB),          256, 0, stream>>>(ctxp, out_ctx);
}
